// Encoder_32212254720642
// MI455X (gfx1250) — 8-bit matrix operand order repaired and checked by reading, compiled, not run on hardware
//
#include <hip/hip_runtime.h>

typedef float v2f __attribute__((ext_vector_type(2)));
typedef float v8f __attribute__((ext_vector_type(8)));
typedef int   v8i __attribute__((ext_vector_type(8)));

#define DIM 256

__device__ __forceinline__ float block_reduce_sum(float v, float* red) {
  int t = threadIdx.x;
  red[t] = v;
  __syncthreads();
  for (int s = 128; s > 0; s >>= 1) {
    if (t < s) red[t] += red[t + s];
    __syncthreads();
  }
  float r = red[0];
  __syncthreads();
  return r;
}

__global__ void norm_rows(const float* __restrict__ in, float* __restrict__ out, int n) {
  __shared__ float red[256];
  int row = blockIdx.x;
  const float* ri = in + (size_t)row * n;
  float s = 0.f;
  for (int c = threadIdx.x; c < n; c += 256) s += ri[c];
  float deg = block_reduce_sum(s, red);
  if (deg == 0.f) deg = 1.f;
  float inv = 1.f / deg;
  float* ro = out + (size_t)row * n;
  for (int c = threadIdx.x; c < n; c += 256) ro[c] = ri[c] * inv;
}

template <int EPI>
__global__ void gemm_f32_wmma(const float* __restrict__ A, const float* __restrict__ B,
                              const float* __restrict__ bias, float* __restrict__ C,
                              int M, int N, int K) {
  int lane = threadIdx.x & 31;
  int wave = threadIdx.x >> 5;
  int tilesM = (M + 15) >> 4;
  int tilesN4 = N >> 6;
  int tile = blockIdx.x * (blockDim.x >> 5) + wave;
  if (tile >= tilesM * tilesN4) return;
  int tm = tile / tilesN4, tn4 = tile % tilesN4;
  int hi = lane >> 4;
  int l15 = lane & 15;

  int rowA = tm * 16 + l15;
  bool rA = rowA < M;
  const float* Arow = A + (size_t)(rA ? rowA : 0) * K;
  int col0 = tn4 * 64 + l15;

  v8f acc0 = {}, acc1 = {}, acc2 = {}, acc3 = {};
  for (int k = 0; k < K; k += 4) {
    int ko = k + hi * 2;
    if (k + 64 < K) __builtin_prefetch(Arow + k + 64, 0, 1);
    v2f a;
    a.x = rA ? Arow[ko] : 0.f;
    a.y = rA ? Arow[ko + 1] : 0.f;
    const float* b0 = B + (size_t)ko * N + col0;
    const float* b1 = b0 + N;
    v2f b;
    b.x = b0[0];  b.y = b1[0];
    acc0 = __builtin_amdgcn_wmma_f32_16x16x4_f32(false, a, false, b, (short)0, acc0, false, false);
    b.x = b0[16]; b.y = b1[16];
    acc1 = __builtin_amdgcn_wmma_f32_16x16x4_f32(false, a, false, b, (short)0, acc1, false, false);
    b.x = b0[32]; b.y = b1[32];
    acc2 = __builtin_amdgcn_wmma_f32_16x16x4_f32(false, a, false, b, (short)0, acc2, false, false);
    b.x = b0[48]; b.y = b1[48];
    acc3 = __builtin_amdgcn_wmma_f32_16x16x4_f32(false, a, false, b, (short)0, acc3, false, false);
  }

  v8f accs[4] = {acc0, acc1, acc2, acc3};
#pragma unroll
  for (int j = 0; j < 4; ++j) {
    int c = tn4 * 64 + 16 * j + l15;
#pragma unroll
    for (int v = 0; v < 8; ++v) {
      int r = tm * 16 + v + hi * 8;
      if (r < M) {
        float x = accs[j][v];
        if (EPI == 1) {
          x += bias[c];
          x = x > 0.f ? x : 0.f;
        }
        C[(size_t)r * N + c] = x;
      }
    }
  }
}

__device__ __forceinline__ v8i load_a_frag_u8(const unsigned int* p, int hi) {
  v8i a;
#pragma unroll
  for (int q = 0; q < 4; ++q) {
    a[2 * q]     = (int)p[4 * q + hi * 2];
    a[2 * q + 1] = (int)p[4 * q + hi * 2 + 1];
  }
  return a;
}
__device__ __forceinline__ v8i load_b_frag_u8(const unsigned int* p, int hi) {
  v8i b;
#pragma unroll
  for (int j = 0; j < 4; ++j) {
    b[2 * j]     = (int)p[4 * j + hi * 2];
    b[2 * j + 1] = (int)p[4 * j + hi * 2 + 1];
  }
  return b;
}

__global__ void bool_gemm_iu8(const unsigned char* __restrict__ A8,
                              const unsigned char* __restrict__ B8T,
                              float* __restrict__ Mout, int kk, int kkp, int np) {
  int lane = threadIdx.x & 31;
  int wave = threadIdx.x >> 5;
  int tiles = kkp >> 5;
  int tile = blockIdx.x * (blockDim.x >> 5) + wave;
  if (tile >= tiles * tiles) return;
  int tm = tile / tiles, tn = tile % tiles;
  int hi = lane >> 4;
  int l15 = lane & 15;

  const unsigned char* arow0 = A8 + (size_t)(tm * 32 + l15) * np;
  const unsigned char* arow1 = arow0 + (size_t)16 * np;
  const unsigned char* brow0 = B8T + (size_t)(tn * 32 + l15) * np;
  const unsigned char* brow1 = brow0 + (size_t)16 * np;

  v8i acc00 = {}, acc01 = {}, acc10 = {}, acc11 = {};
  for (int k = 0; k < np; k += 64) {
    if (k + 256 < np) {
      __builtin_prefetch(arow0 + k + 256, 0, 1);
      __builtin_prefetch(arow1 + k + 256, 0, 1);
      __builtin_prefetch(brow0 + k + 256, 0, 1);
      __builtin_prefetch(brow1 + k + 256, 0, 1);
    }
    v8i a0 = load_a_frag_u8((const unsigned int*)(arow0 + k), hi);
    v8i a1 = load_a_frag_u8((const unsigned int*)(arow1 + k), hi);
    v8i b0 = load_b_frag_u8((const unsigned int*)(brow0 + k), hi);
    v8i b1 = load_b_frag_u8((const unsigned int*)(brow1 + k), hi);
    acc00 = __builtin_amdgcn_wmma_i32_16x16x64_iu8(false, a0, false, b0, acc00, false, false);
    acc01 = __builtin_amdgcn_wmma_i32_16x16x64_iu8(false, a0, false, b1, acc01, false, false);
    acc10 = __builtin_amdgcn_wmma_i32_16x16x64_iu8(false, a1, false, b0, acc10, false, false);
    acc11 = __builtin_amdgcn_wmma_i32_16x16x64_iu8(false, a1, false, b1, acc11, false, false);
  }

#pragma unroll
  for (int v = 0; v < 8; ++v) {
    int r0 = tm * 32 + v + hi * 8;
    int r1 = r0 + 16;
    int c0 = tn * 32 + l15;
    int c1 = c0 + 16;
    if (r0 < kk && c0 < kk) Mout[(size_t)r0 * kk + c0] = (acc00[v] != 0) ? 1.0f : 0.0f;
    if (r0 < kk && c1 < kk) Mout[(size_t)r0 * kk + c1] = (acc01[v] != 0) ? 1.0f : 0.0f;
    if (r1 < kk && c0 < kk) Mout[(size_t)r1 * kk + c0] = (acc10[v] != 0) ? 1.0f : 0.0f;
    if (r1 < kk && c1 < kk) Mout[(size_t)r1 * kk + c1] = (acc11[v] != 0) ? 1.0f : 0.0f;
  }
}

__global__ void add_layernorm(const float* __restrict__ h, const float* __restrict__ h1,
                              const float* __restrict__ gamma, const float* __restrict__ beta,
                              float* __restrict__ out) {
  __shared__ float red[DIM];
  int row = blockIdx.x, c = threadIdx.x;
  float x = h[(size_t)row * DIM + c] + h1[(size_t)row * DIM + c];
  float mu = block_reduce_sum(x, red) * (1.0f / DIM);
  float d = x - mu;
  float var = block_reduce_sum(d * d, red) * (1.0f / DIM);
  out[(size_t)row * DIM + c] = gamma[c] * d * rsqrtf(var + 1e-5f) + beta[c];
}

__global__ void score_rows(const float* __restrict__ h, const float* __restrict__ pw,
                           const float* __restrict__ pb, int li, float* __restrict__ s) {
  __shared__ float red[DIM];
  int row = blockIdx.x, c = threadIdx.x;
  float dot = block_reduce_sum(h[(size_t)row * DIM + c] * pw[c], red);
  if (c == 0) s[row] = 1.0f / (1.0f + __expf(-(dot + pb[li])));
}

__global__ void topk_rank(const float* __restrict__ s, int n, int kk,
                          int* __restrict__ idx_out, float* __restrict__ val_out,
                          float* __restrict__ idxf_out) {
  __shared__ float tile[256];
  int i = blockIdx.x * 256 + threadIdx.x;
  float si = (i < n) ? s[i] : -1e30f;
  int rank = 0;
  for (int base = 0; base < n; base += 256) {
    int j = base + threadIdx.x;
    tile[threadIdx.x] = (j < n) ? s[j] : -1e30f;
    __syncthreads();
    int lim = n - base;
    if (lim > 256) lim = 256;
    for (int jj = 0; jj < lim; ++jj) {
      float sj = tile[jj];
      if (sj > si || (sj == si && (base + jj) < i)) rank++;
    }
    __syncthreads();
  }
  if (i < n && rank < kk) {
    idx_out[rank] = i;
    val_out[rank] = si;
    idxf_out[rank] = (float)i;
  }
}

__global__ void gather_scale_rows(const float* __restrict__ h, const int* __restrict__ idx,
                                  const float* __restrict__ vals, float* __restrict__ out,
                                  int kk) {
  int id = blockIdx.x * blockDim.x + threadIdx.x;
  if (id >= kk * DIM) return;
  int r = id / DIM, c = id % DIM;
  out[id] = h[(size_t)idx[r] * DIM + c] * vals[r];
}

__global__ void gather_rows_u8(const float* __restrict__ gn, const int* __restrict__ idx,
                               unsigned char* __restrict__ A8, int kk, int kkp, int n, int np) {
  size_t id = (size_t)blockIdx.x * blockDim.x + threadIdx.x;
  if (id >= (size_t)kkp * np) return;
  int r = (int)(id / np), k = (int)(id % np);
  unsigned char v = 0;
  if (r < kk && k < n) v = (gn[(size_t)idx[r] * n + k] != 0.0f) ? 1 : 0;
  A8[id] = v;
}

__global__ void gather_cols_u8(const float* __restrict__ gn, const int* __restrict__ idx,
                               unsigned char* __restrict__ B8T, int kk, int kkp, int n, int np) {
  size_t id = (size_t)blockIdx.x * blockDim.x + threadIdx.x;
  if (id >= (size_t)kkp * np) return;
  int c = (int)(id / np), k = (int)(id % np);
  unsigned char v = 0;
  if (c < kk && k < n) v = (gn[(size_t)k * n + idx[c]] != 0.0f) ? 1 : 0;
  B8T[id] = v;
}

static inline int cdiv(long long a, long long b) { return (int)((a + b - 1) / b); }

extern "C" void kernel_launch(void* const* d_in, const int* in_sizes, int n_in,
                              void* d_out, int out_size, void* d_ws, size_t ws_size,
                              hipStream_t stream) {
  const float* g_in  = (const float*)d_in[0];
  const float* h_in  = (const float*)d_in[1];
  const float* Wg    = (const float*)d_in[2];
  const float* bg    = (const float*)d_in[3];
  const float* gamma = (const float*)d_in[4];
  const float* beta  = (const float*)d_in[5];
  const float* pw    = (const float*)d_in[6];
  const float* pb    = (const float*)d_in[7];

  const int N0 = 4096, N1 = 3276, N2 = 1965;
  const int KKP0 = 3296, NP0 = 4096;
  const int KKP1 = 1984, NP1 = 3328;

  float* out = (float*)d_out;
  size_t o = 0;
  float* out_g = out + o;  o += (size_t)N2 * N2;
  float* out_h = out + o;  o += (size_t)N2 * DIM;
  float* adj0  = out + o;  o += (size_t)N0 * N0;
  float* adj1  = out + o;  o += (size_t)N1 * N1;
  float* down0 = out + o;  o += (size_t)N0 * DIM;
  float* down1 = out + o;  o += (size_t)N1 * DIM;
  float* idx0f = out + o;  o += (size_t)N1;
  float* idx1f = out + o;  o += (size_t)N2;

  char* w = (char*)d_ws;
  auto carve = [&](size_t bytes) {
    char* p = w;
    w += (bytes + 255) & ~(size_t)255;
    return p;
  };
  float* t     = (float*)carve((size_t)N0 * DIM * 4);
  float* h1    = (float*)carve((size_t)N0 * DIM * 4);
  float* sc    = (float*)carve((size_t)N0 * 4);
  float* vals  = (float*)carve((size_t)N0 * 4);
  int*   idxI  = (int*)  carve((size_t)N0 * 4);
  float* newh0 = (float*)carve((size_t)N1 * DIM * 4);
  unsigned char* A8  = (unsigned char*)carve((size_t)KKP0 * NP0);
  unsigned char* B8T = (unsigned char*)carve((size_t)KKP0 * NP0);
  float* mbuf  = (float*)carve((size_t)N1 * N1 * 4);

  norm_rows<<<N0, 256, 0, stream>>>(g_in, adj0, N0);

  gemm_f32_wmma<0><<<cdiv((long long)cdiv(N0,16) * (DIM/64), 8), 256, 0, stream>>>(
      adj0, h_in, nullptr, t, N0, DIM, N0);
  gemm_f32_wmma<1><<<cdiv((long long)cdiv(N0,16) * (DIM/64), 8), 256, 0, stream>>>(
      t, Wg, bg, h1, N0, DIM, DIM);

  add_layernorm<<<N0, 256, 0, stream>>>(h_in, h1, gamma, beta, down0);
  score_rows<<<N0, 256, 0, stream>>>(down0, pw, pb, 0, sc);
  topk_rank<<<cdiv(N0, 256), 256, 0, stream>>>(sc, N0, N1, idxI, vals, idx0f);
  gather_scale_rows<<<cdiv((long long)N1 * DIM, 256), 256, 0, stream>>>(down0, idxI, vals,
                                                                        newh0, N1);

  gather_rows_u8<<<cdiv((long long)KKP0 * NP0, 256), 256, 0, stream>>>(adj0, idxI, A8,
                                                                      N1, KKP0, N0, NP0);
  gather_cols_u8<<<cdiv((long long)KKP0 * NP0, 256), 256, 0, stream>>>(adj0, idxI, B8T,
                                                                      N1, KKP0, N0, NP0);
  {
    int tiles = KKP0 >> 5;
    bool_gemm_iu8<<<cdiv((long long)tiles * tiles, 8), 256, 0, stream>>>(A8, B8T, mbuf,
                                                                         N1, KKP0, NP0);
  }
  norm_rows<<<N1, 256, 0, stream>>>(mbuf, adj1, N1);

  gemm_f32_wmma<0><<<cdiv((long long)cdiv(N1,16) * (DIM/64), 8), 256, 0, stream>>>(
      adj1, newh0, nullptr, t, N1, DIM, N1);
  gemm_f32_wmma<1><<<cdiv((long long)cdiv(N1,16) * (DIM/64), 8), 256, 0, stream>>>(
      t, Wg + DIM * DIM, bg + DIM, h1, N1, DIM, DIM);

  add_layernorm<<<N1, 256, 0, stream>>>(newh0, h1, gamma + DIM, beta + DIM, down1);
  score_rows<<<N1, 256, 0, stream>>>(down1, pw + DIM, pb, 1, sc);
  topk_rank<<<cdiv(N1, 256), 256, 0, stream>>>(sc, N1, N2, idxI, vals, idx1f);
  gather_scale_rows<<<cdiv((long long)N2 * DIM, 256), 256, 0, stream>>>(down1, idxI, vals,
                                                                        out_h, N2);

  gather_rows_u8<<<cdiv((long long)KKP1 * NP1, 256), 256, 0, stream>>>(adj1, idxI, A8,
                                                                      N2, KKP1, N1, NP1);
  gather_cols_u8<<<cdiv((long long)KKP1 * NP1, 256), 256, 0, stream>>>(adj1, idxI, B8T,
                                                                      N2, KKP1, N1, NP1);
  {
    int tiles = KKP1 >> 5;
    bool_gemm_iu8<<<cdiv((long long)tiles * tiles, 8), 256, 0, stream>>>(A8, B8T, mbuf,
                                                                         N2, KKP1, NP1);
  }
  norm_rows<<<N2, 256, 0, stream>>>(mbuf, out_g, N2);
}
